// Prior_MemoryEncoder_68410239090768
// MI455X (gfx1250) — matrix operand order repaired and checked by reading, not compiled, not run on hardware
//
#include <hip/hip_runtime.h>
#include <hip/hip_bf16.h>

typedef __bf16 bf16;
typedef __attribute__((ext_vector_type(16))) __bf16 v16bf;
typedef __attribute__((ext_vector_type(8)))  __bf16 v8bf;
typedef __attribute__((ext_vector_type(4)))  __bf16 v4bf;
typedef __attribute__((ext_vector_type(8)))  float  v8f;
typedef __attribute__((ext_vector_type(4)))  float  v4f;
typedef __attribute__((ext_vector_type(4)))  unsigned int u32x4;
typedef __attribute__((ext_vector_type(8)))  int i32x8;
typedef __attribute__((ext_vector_type(4)))  int i32x4;

#define BN_EPS 1e-5f

#if __has_builtin(__builtin_amdgcn_tensor_load_to_lds) && __has_builtin(__builtin_amdgcn_s_wait_tensorcnt)
#define HAVE_TDM 1
#else
#define HAVE_TDM 0
#endif

__device__ __forceinline__ int iclamp(int v, int hi) { return v < hi ? v : hi; }

// ---------------------------------------------------------------------------
// CDNA5 wave32 WMMA 16x16x32 bf16 fragment layouts:
// A 16x32 (MxK): M = lane&15; elem e -> K = khi + e (contiguous, order used here)
//   (khi = 16 for lanes>=16)
// B 32x16 (KxN): N = lane&15; elem e -> K = e + (lane>=16 ? 16 : 0) (this code's order, matches A)
// C/D f32 16x16: elem r -> M = r + (lane>=16 ? 8 : 0), N = lane&15
// ---------------------------------------------------------------------------

// Unguarded strided loaders (operands are padded; 32-bit address math).
__device__ __forceinline__ v16bf frag_a_ng(const bf16* __restrict__ A, int rs, int cs,
                                           int m0, int k0, int lane) {
  v16bf f;
  int m = m0 + (lane & 15);
  int khi = (lane >= 16) ? 16 : 0;
  int base = m * rs + (k0 + khi) * cs;
#pragma unroll
  for (int e = 0; e < 16; ++e) {
    int kofs = e;
    f[e] = A[base + kofs * cs];
  }
  return f;
}

__device__ __forceinline__ v16bf frag_b_ng(const bf16* __restrict__ B, int rs, int cs,
                                           int k0, int n0, int lane) {
  v16bf f;
  int n = n0 + (lane & 15);
  int khi = (lane >= 16) ? 16 : 0;
  int base = (k0 + khi) * rs + n * cs;
#pragma unroll
  for (int e = 0; e < 16; ++e) f[e] = B[base + e * rs];
  return f;
}

// Fast loaders: contiguous-K, 2x 16-byte vector loads per fragment (no guards;
// call sites guarantee M%16==0 / padded B rows, 16B alignment, k0+32<=K).
__device__ __forceinline__ v16bf frag_a_fast(const bf16* __restrict__ A, long rs,
                                             int m0, int k0, int lane) {
  int m = m0 + (lane & 15);
  int khi = (lane >= 16) ? 16 : 0;
  const bf16* base = A + (long)m * rs + k0 + khi;
  v8bf p0 = *(const v8bf*)(base);
  v8bf p1 = *(const v8bf*)(base + 8);
  return __builtin_shufflevector(p0, p1, 0,1,2,3,4,5,6,7,8,9,10,11,12,13,14,15);
}

__device__ __forceinline__ v16bf frag_b_fast(const bf16* __restrict__ B, long cs,
                                             int k0, int n0, int lane) {
  int n = n0 + (lane & 15);
  int khi = (lane >= 16) ? 16 : 0;
  const bf16* base = B + (long)n * cs + k0 + khi;
  v8bf p0 = *(const v8bf*)(base);
  v8bf p1 = *(const v8bf*)(base + 8);
  return __builtin_shufflevector(p0, p1, 0,1,2,3,4,5,6,7,8,9,10,11,12,13,14,15);
}

// ---------------------------------------------------------------------------
// Fast GEMM: C[M,N] = A[M,K](row-major) * B^T-style (B[n*b_cs + k]) + bias
// K multiple of 32, M multiple of 16. One wave per 16x64 tile.
// ---------------------------------------------------------------------------
template <bool FULLN>
__global__ __launch_bounds__(32)
void k_gemm_fast(const bf16* __restrict__ A, long a_rs,
                 const bf16* __restrict__ B, long b_cs,
                 const float* __restrict__ bias,
                 float* __restrict__ Cf, bf16* __restrict__ Cb, long c_rs,
                 int N, int K) {
  int lane = threadIdx.x;
  int m0 = blockIdx.x * 16;
  int n0base = blockIdx.y * 64;

  v8f acc[4];
#pragma unroll
  for (int s = 0; s < 4; ++s)
#pragma unroll
    for (int r = 0; r < 8; ++r) acc[s][r] = 0.0f;

  for (int k0 = 0; k0 < K; k0 += 32) {
    if (k0 + 32 < K)
      __builtin_prefetch(&A[(long)(m0 + (lane & 15)) * a_rs + (k0 + 32)], 0, 1);
    v16bf af = frag_a_fast(A, a_rs, m0, k0, lane);
#pragma unroll
    for (int s = 0; s < 4; ++s) {
      if (FULLN || (n0base + 16 * s) < N) {  // uniform branch
        v16bf bfv = frag_b_fast(B, b_cs, k0, n0base + 16 * s, lane);
        acc[s] = __builtin_amdgcn_wmma_f32_16x16x32_bf16(
            false, af, false, bfv, (short)0, acc[s], false, false);
      }
    }
  }

  int moff = (lane >= 16) ? 8 : 0;
  int ncol = lane & 15;
#pragma unroll
  for (int s = 0; s < 4; ++s) {
    int n = n0base + 16 * s + ncol;
    if (!FULLN && n >= N) continue;
    float bv = bias ? bias[n] : 0.0f;
#pragma unroll
    for (int r = 0; r < 8; ++r) {
      int m = m0 + r + moff;
      float y = acc[s][r] + bv;
      long off = (long)m * c_rs + n;
      if (Cf) Cf[off] = y;
      if (Cb) Cb[off] = (bf16)y;
    }
  }
}

// Small GEMM: fully unguarded (padded operands), 16x16 per wave, int math.
__global__ __launch_bounds__(32)
void k_gemm_small(const bf16* __restrict__ A, int a_rs, int a_cs,
                  const bf16* __restrict__ B, int b_rs, int b_cs,
                  const float* __restrict__ bias, int biasN,
                  float* __restrict__ Cf, bf16* __restrict__ Cb, int c_rs,
                  int K) {
  int lane = threadIdx.x;
  int m0 = blockIdx.x * 16;

  v8f acc;
#pragma unroll
  for (int r = 0; r < 8; ++r) acc[r] = 0.0f;

  for (int k0 = 0; k0 < K; k0 += 32) {
    v16bf af = frag_a_ng(A, a_rs, a_cs, m0, k0, lane);
    v16bf bfv = frag_b_ng(B, b_rs, b_cs, k0, 0, lane);
    acc = __builtin_amdgcn_wmma_f32_16x16x32_bf16(
        false, af, false, bfv, (short)0, acc, false, false);
  }

  int moff = (lane >= 16) ? 8 : 0;
  int n = lane & 15;
  float bv = bias ? bias[iclamp(n, biasN - 1)] : 0.0f;
#pragma unroll
  for (int r = 0; r < 8; ++r) {
    int m = m0 + r + moff;
    float y = acc[r] + bv;
    int off = m * c_rs + n;
    if (Cf) Cf[off] = y;
    if (Cb) Cb[off] = (bf16)y;
  }
}

// ---------------------------------------------------------------------------
// Conv1d(k=3,p=1) + bias/ReLU/BN, LDS-staged via Tensor Data Mover.
// grid = (48 pose-tiles, 1, 256 batches), block = 128 (4 waves).
// ---------------------------------------------------------------------------
__global__ __launch_bounds__(128)
void k_conv_bn(const bf16* __restrict__ Aw, int a_ld, int Kc, int Cin,
               const bf16* __restrict__ Xb, long x_bs,
               const float* __restrict__ bias,
               const float* __restrict__ bn_g, const float* __restrict__ bn_b,
               const float* __restrict__ bn_m, const float* __restrict__ bn_v,
               bf16* __restrict__ Pb, long p_bs) {
  __shared__ bf16 slab[180 * 32];
  int n0 = blockIdx.x * 16;
  int b = blockIdx.z;
  const bf16* X = Xb + (long)b * x_bs;

  bool interior = (n0 >= 8) && (n0 + 24 <= 768);
#if HAVE_TDM
  if (interior) {
    if (threadIdx.x < 32) {  // wave 0 issues one TDM descriptor (EXEC ignored by TDM)
      unsigned long long ga = (unsigned long long)(const void*)(X + (n0 - 8));
      unsigned int lds_off = (unsigned int)(unsigned long long)(const void*)&slab[0];
      u32x4 g0;
      g0[0] = 1u;                                             // count=1, user mode
      g0[1] = lds_off;                                        // lds_addr (bytes)
      g0[2] = (unsigned int)(ga & 0xffffffffull);             // global_addr[31:0]
      g0[3] = (unsigned int)((ga >> 32) & 0x1ffffffull) | (2u << 30);  // addr[56:32]|type=2
      i32x8 g1;
      g1[0] = 0x00010000;                                     // data_size=2B
      g1[1] = (int)(768u << 16);                              // tensor_dim0=768 (lo16)
      g1[2] = (int)((unsigned)Cin << 16);                     // dim0 hi=0 | tensor_dim1=Cin
      g1[3] = (int)(32u << 16);                               // dim1 hi=0 | tile_dim0=32
      g1[4] = Cin;                                            // tile_dim1=Cin, tile_dim2=0
      g1[5] = 768;                                            // tensor_dim0_stride lo32
      g1[6] = 0;
      g1[7] = 0;
      i32x4 g2 = {0, 0, 0, 0}, g3 = {0, 0, 0, 0};
      i32x8 g4 = {0, 0, 0, 0, 0, 0, 0, 0};
      __builtin_amdgcn_tensor_load_to_lds(g0, g1, g2, g3, g4, 0);
      __builtin_amdgcn_s_wait_tensorcnt(0);
    }
  } else
#endif
  {
    for (int idx = threadIdx.x; idx < Cin * 32; idx += (int)blockDim.x) {
      int i = idx >> 5, c = idx & 31;
      int pos = n0 - 8 + c;
      bf16 v = (bf16)0.0f;
      if (pos >= 0 && pos < 768) v = X[(long)i * 768 + pos];
      slab[idx] = v;
    }
  }
  __syncthreads();

  int wave = threadIdx.x >> 5;
  int lane = threadIdx.x & 31;
  int khi16 = (lane >= 16) ? 16 : 0;
  int lc = (lane & 15) + 7;  // slab-local column for t=0

#pragma unroll
  for (int q = 0; q < 3; ++q) {
    int m0 = (wave * 3 + q) * 16;
    v8f acc;
#pragma unroll
    for (int r = 0; r < 8; ++r) acc[r] = 0.0f;

    for (int k0 = 0; k0 < Kc; k0 += 32) {
      v16bf af = frag_a_fast(Aw, a_ld, m0, k0, lane);
      v16bf bfv;
#pragma unroll
      for (int e = 0; e < 16; ++e) {
        // clamp only; A-panel columns >= Kc are zero, killing any residue
        int kc = iclamp(k0 + e + khi16, Kc - 1);
        int i = kc / 3;
        int t = kc - 3 * i;
        bfv[e] = slab[i * 32 + lc + t];
      }
      acc = __builtin_amdgcn_wmma_f32_16x16x32_bf16(
          false, af, false, bfv, (short)0, acc, false, false);
    }

    int moff = (lane >= 16) ? 8 : 0;
    int n = n0 + (lane & 15);
#pragma unroll
    for (int r = 0; r < 8; ++r) {
      int o = m0 + r + moff;
      if (o < 180) {
        float y = acc[r] + bias[o];
        y = fmaxf(y, 0.0f);
        y = (y - bn_m[o]) * __frsqrt_rn(bn_v[o] + BN_EPS) * bn_g[o] + bn_b[o];
        Pb[(long)b * p_bs + (long)o * 768 + n] = (bf16)y;
      }
    }
  }
}

// --------------------------- small helper kernels ---------------------------
__global__ void k_f2b4(const float* __restrict__ in, bf16* __restrict__ out, long n4) {
  long i = (long)blockIdx.x * blockDim.x + threadIdx.x;
  if (i >= n4) return;
  v4f x4 = ((const v4f*)in)[i];
  v4bf o;
#pragma unroll
  for (int j = 0; j < 4; ++j) o[j] = (bf16)x4[j];
  ((v4bf*)out)[i] = o;
}

// Zero-padded bf16 panel [rows x ld] from f32 src [R x C]
__global__ void k_pad(const float* __restrict__ src, bf16* __restrict__ dst,
                      int R, int C, int ld, long total) {
  long idx = (long)blockIdx.x * blockDim.x + threadIdx.x;
  if (idx >= total) return;
  int r = (int)(idx / ld);
  int c = (int)(idx % ld);
  float v = (r < R && c < C) ? src[(long)r * C + c] : 0.0f;
  dst[idx] = (bf16)v;
}

// SP gate: score=dot(mem[b], p[b,c,:]); sig=sigmoid; p = sig*p + (1-sig)*mem
__global__ void k_sp_gate(const float* __restrict__ memf, bf16* __restrict__ Pb, long p_bs) {
  int c = blockIdx.x, b = blockIdx.y;
  __shared__ float red[256];
  long base = (long)b * p_bs + (long)c * 768;
  float s = 0.0f;
  for (int j = threadIdx.x; j < 768; j += 256)
    s += (float)Pb[base + j] * memf[(long)b * 768 + j];
  red[threadIdx.x] = s;
  __syncthreads();
  for (int st = 128; st > 0; st >>= 1) {
    if ((int)threadIdx.x < st) red[threadIdx.x] += red[threadIdx.x + st];
    __syncthreads();
  }
  float sig = 1.0f / (1.0f + __expf(-red[0]));
  for (int j = threadIdx.x; j < 768; j += 256) {
    float y = sig * (float)Pb[base + j] + (1.0f - sig) * memf[(long)b * 768 + j];
    Pb[base + j] = (bf16)y;
  }
}

// TM: softmax over the 10 chunk scores (stride 16), scale chunk rows by (1 + soft)
__global__ void k_tm_scale(const float* __restrict__ score2, bf16* __restrict__ Pb, long p_bs) {
  int b = blockIdx.x;
  __shared__ float sm[16];
  if (threadIdx.x == 0) {
    float mx = -1e30f;
    for (int c = 0; c < 10; ++c) mx = fmaxf(mx, score2[b * 16 + c]);
    float sum = 0.0f;
    for (int c = 0; c < 10; ++c) { float e = __expf(score2[b * 16 + c] - mx); sm[c] = e; sum += e; }
    for (int c = 0; c < 10; ++c) sm[c] /= sum;
  }
  __syncthreads();
  for (int t = threadIdx.x; t < 7680; t += blockDim.x) {
    int c = t / 768;
    long off = (long)b * p_bs + t;  // chunk rows are contiguous
    Pb[off] = (bf16)((float)Pb[off] * (1.0f + sm[c]));
  }
}

// cat = [x (60 frames) ; p (180 frames)] per batch, 8 bf16 per thread
__global__ void k_cat8(const v8bf* __restrict__ xb, const v8bf* __restrict__ Pb,
                       v8bf* __restrict__ cat) {
  long idx = (long)blockIdx.x * blockDim.x + threadIdx.x;
  const long TOT8 = (long)256 * 240 * 96;
  if (idx >= TOT8) return;
  long b = idx / (240 * 96);
  long rem = idx - b * (240 * 96);
  int f = (int)(rem / 96);
  int j = (int)(rem - (long)f * 96);
  cat[idx] = (f < 60) ? xb[b * (60 * 96) + (long)f * 96 + j]
                      : Pb[b * (180 * 96) + (long)(f - 60) * 96 + j];
}

// ---------------------------------------------------------------------------
extern "C" void kernel_launch(void* const* d_in, const int* in_sizes, int n_in,
                              void* d_out, int out_size, void* d_ws, size_t ws_size,
                              hipStream_t stream) {
  (void)in_sizes; (void)n_in; (void)out_size; (void)ws_size;

  const float* x        = (const float*)d_in[0];
  const float* conv1_w  = (const float*)d_in[1];
  const float* conv1_b  = (const float*)d_in[2];
  const float* bn1_g = (const float*)d_in[3], *bn1_b = (const float*)d_in[4];
  const float* bn1_m = (const float*)d_in[5], *bn1_v = (const float*)d_in[6];
  const float* conv2_w  = (const float*)d_in[7];
  const float* conv2_b  = (const float*)d_in[8];
  const float* bn2_g = (const float*)d_in[9],  *bn2_b = (const float*)d_in[10];
  const float* bn2_m = (const float*)d_in[11], *bn2_v = (const float*)d_in[12];
  const float* sp_w1 = (const float*)d_in[13], *sp_b1 = (const float*)d_in[14];
  const float* sp_w2 = (const float*)d_in[15], *sp_b2 = (const float*)d_in[16];
  const float* tmc_w1 = (const float*)d_in[17], *tmc_b1 = (const float*)d_in[18];
  const float* tmc_w2 = (const float*)d_in[19], *tmc_b2 = (const float*)d_in[20];
  const float* tmm_w1 = (const float*)d_in[21], *tmm_b1 = (const float*)d_in[22];
  const float* tmm_w2 = (const float*)d_in[23], *tmm_b2 = (const float*)d_in[24];
  const float* post_w1 = (const float*)d_in[25], *post_b1 = (const float*)d_in[26];
  const float* post_w2 = (const float*)d_in[27], *post_b2 = (const float*)d_in[28];
  float* out = (float*)d_out;

  // ---- workspace carve-out ----
  char* cur = (char*)d_ws;
  auto alloc = [&](size_t bytes) -> void* {
    void* r = (void*)cur;
    cur += (bytes + 255) & ~(size_t)255;
    return r;
  };
  const long NX = 256L * 60 * 768;
  const long NP = 256L * 180 * 768;
  const long NC = 256L * 240 * 768;
  bf16* xb      = (bf16*)alloc(NX * 2);
  bf16* Ac1     = (bf16*)alloc(192L * 192 * 2);
  bf16* Ac2     = (bf16*)alloc(192L * 544 * 2);
  bf16* spw1b   = (bf16*)alloc(768L * 7680 * 2);
  bf16* spw2b   = (bf16*)alloc(768L * 768 * 2);
  bf16* tmcw1b  = (bf16*)alloc(768L * 7680 * 2);
  bf16* tmcw2b  = (bf16*)alloc(768L * 768 * 2);
  bf16* tmmw1b  = (bf16*)alloc(16L * 7680 * 2);   // 10 rows used, 6 pad rows
  bf16* tmmw2b  = (bf16*)alloc(32L * 32 * 2);     // zero-padded 10x10 panel
  bf16* pw1b    = (bf16*)alloc(768L * 768 * 2);
  bf16* pw2b    = (bf16*)alloc(768L * 768 * 2);
  bf16* p1b     = (bf16*)alloc(NP * 2);
  bf16* p2b     = (bf16*)alloc(NP * 2);
  bf16* hspb    = (bf16*)alloc(256L * 768 * 2);
  float* memf   = (float*)alloc(256L * 768 * 4);
  bf16* htmcb   = (bf16*)alloc(256L * 768 * 2);
  bf16* mem2b   = (bf16*)alloc(256L * 768 * 2);
  bf16* t1b     = (bf16*)alloc(256L * 32 * 2);    // ld=32, cols>=10 pad
  bf16* pencb   = (bf16*)alloc(256L * 16 * 2);    // ld=16
  bf16* Gb      = (bf16*)alloc(768L * 16 * 2);    // ld=16
  float* score2 = (float*)alloc(256L * 16 * 4);   // ld=16
  bf16* catb    = (bf16*)alloc(NC * 2);
  bf16* midb    = (bf16*)alloc(NC * 2);

  auto f2b = [&](const float* src, bf16* dst, long n) {
    long n4 = n / 4;
    k_f2b4<<<dim3((unsigned)((n4 + 255) / 256)), dim3(256), 0, stream>>>(src, dst, n4);
  };

  // ---- stage 0: precision conversion / weight prep ----
  f2b(x, xb, NX);
  k_pad<<<dim3((192 * 192 + 255) / 256), dim3(256), 0, stream>>>(conv1_w, Ac1, 180, 180, 192, 192L * 192);
  k_pad<<<dim3((192 * 544 + 255) / 256), dim3(256), 0, stream>>>(conv2_w, Ac2, 180, 540, 544, 192L * 544);
  k_pad<<<dim3((32 * 32 + 255) / 256), dim3(256), 0, stream>>>(tmm_w2, tmmw2b, 10, 10, 32, 32L * 32);
  f2b(sp_w1, spw1b, 768L * 7680);
  f2b(sp_w2, spw2b, 768L * 768);
  f2b(tmc_w1, tmcw1b, 768L * 7680);
  f2b(tmc_w2, tmcw2b, 768L * 768);
  f2b(tmm_w1, tmmw1b, 10L * 7680);
  f2b(post_w1, pw1b, 768L * 768);
  f2b(post_w2, pw2b, 768L * 768);

  // ---- stage 1: pred_conv (conv+relu+bn, twice), TDM-staged LDS ----
  k_conv_bn<<<dim3(48, 1, 256), dim3(128), 0, stream>>>(
      Ac1, 192, 180, 60, xb, 46080, conv1_b, bn1_g, bn1_b, bn1_m, bn1_v, p1b, 138240);
  k_conv_bn<<<dim3(48, 1, 256), dim3(128), 0, stream>>>(
      Ac2, 544, 540, 180, p1b, 138240, conv2_b, bn2_g, bn2_b, bn2_m, bn2_v, p2b, 138240);

  // tail = x[:, 50:, :] flattened -> rows of xb with offset 50*768, stride 46080
  const bf16* tailb = xb + 50L * 768;

  // ---- stage 2: SP memory net ----
  k_gemm_fast<true><<<dim3(16, 12), dim3(32), 0, stream>>>(
      tailb, 46080, spw1b, 7680, sp_b1, nullptr, hspb, 768, 768, 7680);
  k_gemm_fast<true><<<dim3(16, 12), dim3(32), 0, stream>>>(
      hspb, 768, spw2b, 768, sp_b2, memf, nullptr, 768, 768, 768);
  k_sp_gate<<<dim3(10, 256), dim3(256), 0, stream>>>(memf, p2b, 138240);

  // ---- stage 3: TM memory net ----
  k_gemm_fast<true><<<dim3(16, 12), dim3(32), 0, stream>>>(
      tailb, 46080, tmcw1b, 7680, tmc_b1, nullptr, htmcb, 768, 768, 7680);
  k_gemm_fast<true><<<dim3(16, 12), dim3(32), 0, stream>>>(
      htmcb, 768, tmcw2b, 768, tmc_b2, nullptr, mem2b, 768, 768, 768);
  // t1 = p_chunk_flat @ tmm_w1^T + b1  (M=256, N=10, K=7680), out ld=32
  k_gemm_fast<false><<<dim3(16, 1), dim3(32), 0, stream>>>(
      p2b, 138240, tmmw1b, 7680, tmm_b1, nullptr, t1b, 32, 10, 7680);
  // penc = t1 @ tmm_w2^T + b2  (K=10 padded to 32; zero panel rows kill pad)
  k_gemm_small<<<dim3(16, 1), dim3(32), 0, stream>>>(
      t1b, 32, 1, tmmw2b, 1, 32, tmm_b2, 10, nullptr, pencb, 16, 32);
  // G = mem2^T @ penc  (M=768, N<=16, K=256)
  k_gemm_small<<<dim3(48, 1), dim3(32), 0, stream>>>(
      mem2b, 1, 768, pencb, 16, 1, nullptr, 0, nullptr, Gb, 16, 256);
  // score2 = mem2 @ G  (M=256, N<=16, K=768)
  k_gemm_small<<<dim3(16, 1), dim3(32), 0, stream>>>(
      mem2b, 768, 1, Gb, 16, 1, nullptr, 0, score2, nullptr, 16, 768);
  k_tm_scale<<<dim3(256), dim3(256), 0, stream>>>(score2, p2b, 138240);

  // ---- stage 4: concat + post_header ----
  k_cat8<<<dim3((unsigned)((256L * 240 * 96 + 255) / 256)), dim3(256), 0, stream>>>(
      (const v8bf*)xb, (const v8bf*)p2b, (v8bf*)catb);
  k_gemm_fast<true><<<dim3(3840, 12), dim3(32), 0, stream>>>(
      catb, 768, pw1b, 768, post_b1, nullptr, midb, 768, 768, 768);
  k_gemm_fast<true><<<dim3(3840, 12), dim3(32), 0, stream>>>(
      midb, 768, pw2b, 768, post_b2, out, nullptr, 768, 768, 768);
}
